// Attention_69758858822125
// MI455X (gfx1250) — matrix operand order repaired and checked by reading, not compiled, not run on hardware
//
#include <hip/hip_runtime.h>

// ---------------------------------------------------------------------------
// Problem constants (B=1): x [C=512, N=4096], heads=8, dim_head=64
// ---------------------------------------------------------------------------
#define NTOK  4096
#define CCH   512
#define NHEAD 8
#define DHEAD 64
#define NGRP  32
#define GCH   16
#define ATTN_SCALE 0.125f   // 64^-0.5

typedef __attribute__((ext_vector_type(16))) _Float16 v16h;
typedef __attribute__((ext_vector_type(8)))  _Float16 v8h;
typedef __attribute__((ext_vector_type(8)))  float    v8f;
typedef __attribute__((ext_vector_type(4)))  unsigned u32x4;
typedef __attribute__((ext_vector_type(8)))  unsigned u32x8;

__device__ __forceinline__ v8f wmma_f16(v16h a, v16h b, v8f c) {
  return __builtin_amdgcn_wmma_f32_16x16x32_f16(
      false, a, false, b, (short)0, c, false, false);
}

// low 32 bits of a generic pointer to LDS == LDS byte offset (flat aperture)
__device__ __forceinline__ unsigned lds_off(const void* p) {
  return (unsigned)(unsigned long long)p;
}

// Tensor Data Mover: 2D f16 tile (tile0 elems/row x tile1 rows, row stride
// stride0 elems) from global -> contiguous LDS. D# per CDNA5 ISA ch.8.
// Issued once per wave (EXEC ignored); tracked by TENSORcnt.
__device__ __forceinline__ void tdm_load_2d_f16(
    unsigned lds_byte_addr, const _Float16* gptr,
    unsigned tile0, unsigned tile1, unsigned stride0) {
  unsigned long long ga = (unsigned long long)gptr;
  u32x4 g0;
  g0[0] = 1u;                                           // count=1 (valid, user)
  g0[1] = lds_byte_addr;                                // lds_addr
  g0[2] = (unsigned)ga;                                 // global_addr[31:0]
  g0[3] = (unsigned)((ga >> 32) & 0x01FFFFFFu)          // global_addr[56:32]
          | (2u << 30);                                 // type=2 (image)
  u32x8 g1;
  g1[0] = 1u << 16;                                     // data_size=1 (2 bytes)
  g1[1] = (tile0 & 0xFFFFu) << 16;                      // tensor_dim0[15:0]
  g1[2] = (tile0 >> 16) | ((tile1 & 0xFFFFu) << 16);    // t_dim0[hi]|t_dim1[lo]
  g1[3] = (tile1 >> 16) | (tile0 << 16);                // t_dim1[hi]|tile_dim0
  g1[4] = tile1 & 0xFFFFu;                              // tile_dim1, tile_dim2=0
  g1[5] = stride0;                                      // tensor_dim0_stride
  g1[6] = 0u;
  g1[7] = 0u;
  asm volatile("tensor_load_to_lds %0, %1" :: "s"(g0), "s"(g1) : "memory");
}

// ---------------------------------------------------------------------------
// 1) f32 -> f16 conversion (weights)
// ---------------------------------------------------------------------------
__global__ void cvt_f32_to_f16(const float* __restrict__ src,
                               _Float16* __restrict__ dst, int n) {
  int i = blockIdx.x * blockDim.x + threadIdx.x;
  if (i < n) dst[i] = (_Float16)src[i];
}

// ---------------------------------------------------------------------------
// 2) GroupNorm over [C,N] + transpose to f16 [N,C]
// ---------------------------------------------------------------------------
__global__ __launch_bounds__(256) void groupnorm_t16(
    const float* __restrict__ x, const float* __restrict__ gamma,
    const float* __restrict__ beta, _Float16* __restrict__ Xn) {
  __shared__ float red1[256];
  __shared__ float red2[256];
  const int c0 = blockIdx.x * GCH;
  const int tid = threadIdx.x;
  const int total = GCH * NTOK;

  float s1 = 0.f, s2 = 0.f;
  for (int i = tid; i < total; i += 256) {
    float v = x[c0 * NTOK + i];
    s1 += v; s2 += v * v;
  }
  red1[tid] = s1; red2[tid] = s2;
  __syncthreads();
  for (int st = 128; st > 0; st >>= 1) {
    if (tid < st) { red1[tid] += red1[tid + st]; red2[tid] += red2[tid + st]; }
    __syncthreads();
  }
  const float inv_n = 1.f / (float)total;
  const float mean = red1[0] * inv_n;
  const float var  = red2[0] * inv_n - mean * mean;
  const float rstd = rsqrtf(var + 1e-5f);

  for (int i = tid; i < total; i += 256) {
    int cl = i >> 12;
    int n  = i & (NTOK - 1);
    int c  = c0 + cl;
    float v = x[c0 * NTOK + i];
    Xn[n * CCH + c] = (_Float16)((v - mean) * rstd * gamma[c] + beta[c]);
  }
}

// ---------------------------------------------------------------------------
// 3) Pipelined WMMA GEMM  C[M,Nout] = A[M,K] @ W[Nout,K]^T
//    256 thr (8 waves); tile 256(M) x 64(N) x 32(K); per wave 32x64 -> 8 WMMA
//    A tile: TDM double-buffered; W tile: register double-buffered
// ---------------------------------------------------------------------------
#define TM 256
#define TN 64
#define TK 32

__global__ __launch_bounds__(256) void gemm_xwT(
    const _Float16* __restrict__ A, const _Float16* __restrict__ W,
    _Float16* __restrict__ Cout, int M, int K, int Nout) {
  __shared__ __align__(32) _Float16 Al[2][TM][TK];   // 32 KB
  __shared__ __align__(32) _Float16 Wt[2][TK][TN];   // 8 KB

  const int tid  = threadIdx.x;
  const int wave = tid >> 5;
  const int lane = tid & 31;
  const int m0 = blockIdx.x * TM;
  const int n0 = blockIdx.y * TN;

  const int arow = lane & 15;
  const int kb   = (lane >> 4) * 8;
  const int wn = tid >> 2;            // W stage: n row 0..63
  const int wkc = (tid & 3) * 8;      // W stage: k chunk

  v8f acc[2][4] = {};
  const int NIT = K / TK;

  // ---- prologue: stage k-step 0 into buffer 0 ----
  {
    v8h wr = *(const v8h*)&W[(size_t)(n0 + wn) * K + wkc];
    if (wave == 0)
      tdm_load_2d_f16(lds_off(&Al[0][0][0]), &A[(size_t)m0 * K], TK, TM, K);
#pragma unroll
    for (int q = 0; q < 8; ++q) Wt[0][wkc + q][wn] = wr[q];
    if (wave == 0) __builtin_amdgcn_s_wait_tensorcnt(0);
  }
  __syncthreads();

  for (int it = 0; it < NIT; ++it) {
    const int cur = it & 1, nxt = cur ^ 1;
    const bool more = (it + 1) < NIT;
    const int kn = (it + 1) * TK;

    // issue next-tile traffic before compute
    v8h wrn;
    if (more) {
      wrn = *(const v8h*)&W[(size_t)(n0 + wn) * K + kn + wkc];
      if (wave == 0)
        tdm_load_2d_f16(lds_off(&Al[nxt][0][0]), &A[(size_t)m0 * K + kn],
                        TK, TM, K);
    }

    // ---- compute on current buffers: 2 A-frags x 4 B-frags = 8 WMMA ----
    union { v16h v; v8h h[2]; } af[2];
#pragma unroll
    for (int f = 0; f < 2; ++f) {
      af[f].h[0] = *(const v8h*)&Al[cur][wave * 32 + f * 16 + arow][kb];
      af[f].h[1] = *(const v8h*)&Al[cur][wave * 32 + f * 16 + arow][kb + 16];
    }
    v16h bf[4];
#pragma unroll
    for (int c = 0; c < 4; ++c)
#pragma unroll
      for (int s = 0; s < 16; ++s)
        bf[c][s] = Wt[cur][kb + (s & 7) + ((s >> 3) << 4)][c * 16 + arow];
#pragma unroll
    for (int f = 0; f < 2; ++f)
#pragma unroll
      for (int c = 0; c < 4; ++c)
        acc[f][c] = wmma_f16(af[f].v, bf[c], acc[f][c]);

    // drain next-tile staging
    if (more) {
#pragma unroll
      for (int q = 0; q < 8; ++q) Wt[nxt][wkc + q][wn] = wrn[q];
      if (wave == 0) __builtin_amdgcn_s_wait_tensorcnt(0);
    }
    __syncthreads();
  }

  // C layout: VGPR r, lanes 0-15: M=r,N=lane; lanes 16-31: M=8+r,N=lane-16
  const int rb = m0 + wave * 32 + ((lane >> 4) << 3);
  const int ci = lane & 15;
#pragma unroll
  for (int f = 0; f < 2; ++f)
#pragma unroll
    for (int c = 0; c < 4; ++c)
#pragma unroll
      for (int r = 0; r < 8; ++r)
        Cout[(size_t)(rb + f * 16 + r) * Nout + n0 + c * 16 + ci] =
            (_Float16)acc[f][c][r];
}

// ---------------------------------------------------------------------------
// 4) Flash attention: block = (64 q-rows, head), 4 waves x 16 rows
//    V tile via TDM (double-buffered), K tile register-pipelined + transposed
// ---------------------------------------------------------------------------
__global__ __launch_bounds__(128) void flash_attn(
    const _Float16* __restrict__ Q, const _Float16* __restrict__ Km,
    const _Float16* __restrict__ V, _Float16* __restrict__ O) {
  __shared__ __align__(32) _Float16 Kt[2][DHEAD][32];  // Kt[b][d][j], 8 KB
  __shared__ __align__(32) _Float16 Vl[2][32][DHEAD];  // V[b][j][d], 8 KB
  __shared__ __align__(32) _Float16 Pl[4][16][32];     // wave-private bounce

  const int tid  = threadIdx.x;
  const int wave = tid >> 5;
  const int lane = tid & 31;
  const int hc   = blockIdx.y * DHEAD;
  const int m0   = blockIdx.x * 64 + wave * 16;

  const int arow = lane & 15;
  const int kb   = (lane >> 4) * 8;

  // loop-invariant Q fragments (d 0..31, 32..63)
  union { v16h v; v8h h2[2]; } aq[2];
#pragma unroll
  for (int f = 0; f < 2; ++f) {
    const _Float16* qp = &Q[(size_t)(m0 + arow) * CCH + hc + f * 32];
    aq[f].h2[0] = *(const v8h*)&qp[kb];
    aq[f].h2[1] = *(const v8h*)&qp[kb + 16];
  }

  float mrow[8], lrow[8];
#pragma unroll
  for (int r = 0; r < 8; ++r) { mrow[r] = -1e30f; lrow[r] = 0.f; }
  v8f oacc[4] = {};

  // K stage helpers: 2 chunks/thread over 32x64 halves
  const int kj0 = tid >> 3;             // chunk t=0: key row
  const int kd0 = (tid & 7) * 8;        // d chunk
  const int kj1 = (tid + 128) >> 3;
  const int kd1 = ((tid + 128) & 7) * 8;

  // ---- prologue: stage tile 0 ----
  {
    v8h k0 = *(const v8h*)&Km[(size_t)kj0 * CCH + hc + kd0];
    v8h k1 = *(const v8h*)&Km[(size_t)kj1 * CCH + hc + kd1];
    if (wave == 0)
      tdm_load_2d_f16(lds_off(&Vl[0][0][0]), &V[hc], DHEAD, 32, CCH);
#pragma unroll
    for (int q = 0; q < 8; ++q) {
      Kt[0][kd0 + q][kj0] = k0[q];
      Kt[0][kd1 + q][kj1] = k1[q];
    }
    if (wave == 0) __builtin_amdgcn_s_wait_tensorcnt(0);
  }
  __syncthreads();

  for (int j0 = 0; j0 < NTOK; j0 += 32) {
    const int cur = (j0 >> 5) & 1, nxt = cur ^ 1;
    const bool more = (j0 + 32) < NTOK;

    v8h kn0, kn1;
    if (more) {
      kn0 = *(const v8h*)&Km[(size_t)(j0 + 32 + kj0) * CCH + hc + kd0];
      kn1 = *(const v8h*)&Km[(size_t)(j0 + 32 + kj1) * CCH + hc + kd1];
      if (wave == 0)
        tdm_load_2d_f16(lds_off(&Vl[nxt][0][0]), &V[(size_t)(j0 + 32) * CCH + hc],
                        DHEAD, 32, CCH);
      if (j0 + 64 < NTOK)   // pull the tile after next toward L2
        __builtin_prefetch(&Km[(size_t)(j0 + 64 + kj0) * CCH + hc], 0, 1);
    }

    // ---- S = Q K^T for 32 keys: two 16x16 tiles ----
    v8f s[2];
#pragma unroll
    for (int jt = 0; jt < 2; ++jt) {
      v16h b0, b1;
#pragma unroll
      for (int s = 0; s < 16; ++s) {
        b0[s] = Kt[cur][kb + (s & 7) + ((s >> 3) << 4)][jt * 16 + arow];
        b1[s] = Kt[cur][32 + kb + (s & 7) + ((s >> 3) << 4)][jt * 16 + arow];
      }
      v8f c = {};
      c = wmma_f16(aq[0].v, b0, c);
      c = wmma_f16(aq[1].v, b1, c);
      s[jt] = c;
    }

    // ---- online softmax (row reductions stay in each 16-lane half) ----
#pragma unroll
    for (int r = 0; r < 8; ++r) {
      float v0 = s[0][r] * ATTN_SCALE;
      float v1 = s[1][r] * ATTN_SCALE;
      float mx = fmaxf(v0, v1);
#pragma unroll
      for (int off = 8; off >= 1; off >>= 1)
        mx = fmaxf(mx, __shfl_xor(mx, off, 32));
      float mnew  = fmaxf(mrow[r], mx);
      float alpha = __expf(mrow[r] - mnew);
      float p0 = __expf(v0 - mnew);
      float p1 = __expf(v1 - mnew);
      float ps = p0 + p1;
#pragma unroll
      for (int off = 8; off >= 1; off >>= 1)
        ps += __shfl_xor(ps, off, 32);
      lrow[r] = lrow[r] * alpha + ps;
      mrow[r] = mnew;
      s[0][r] = p0; s[1][r] = p1;
#pragma unroll
      for (int c = 0; c < 4; ++c) oacc[c][r] *= alpha;
    }

    // ---- P: C-layout -> LDS -> A-layout (wave-private region) ----
    {
      const int prow = (lane >> 4) << 3;
      const int pc   = lane & 15;
#pragma unroll
      for (int r = 0; r < 8; ++r) {
        Pl[wave][prow + r][pc]      = (_Float16)s[0][r];
        Pl[wave][prow + r][pc + 16] = (_Float16)s[1][r];
      }
    }
    __asm__ volatile("s_wait_dscnt 0" ::: "memory");
    union { v16h v; v8h h2[2]; } ap;
    ap.h2[0] = *(const v8h*)&Pl[wave][arow][kb];
    ap.h2[1] = *(const v8h*)&Pl[wave][arow][kb + 16];

    // ---- O += P(16x32) V(32x64) ----
#pragma unroll
    for (int c = 0; c < 4; ++c) {
      v16h bv;
#pragma unroll
      for (int s = 0; s < 16; ++s)
        bv[s] = Vl[cur][kb + (s & 7) + ((s >> 3) << 4)][c * 16 + arow];
      oacc[c] = wmma_f16(ap.v, bv, oacc[c]);
    }

    // drain next-tile K staging
    if (more) {
#pragma unroll
      for (int q = 0; q < 8; ++q) {
        Kt[nxt][kd0 + q][kj0] = kn0[q];
        Kt[nxt][kd1 + q][kj1] = kn1[q];
      }
      if (wave == 0) __builtin_amdgcn_s_wait_tensorcnt(0);
    }
    __syncthreads();
  }

  // ---- epilogue: normalize, store f16 [N, heads*dhead] ----
  const int rbase = (lane >> 4) << 3;
  const int cidx  = lane & 15;
#pragma unroll
  for (int r = 0; r < 8; ++r) {
    float inv = 1.f / lrow[r];
#pragma unroll
    for (int c = 0; c < 4; ++c)
      O[(size_t)(m0 + rbase + r) * CCH + hc + c * 16 + cidx] =
          (_Float16)(oacc[c][r] * inv);
  }
}

// ---------------------------------------------------------------------------
// 5) Out projection (same pipeline), fused bias + residual, transposed store
// ---------------------------------------------------------------------------
__global__ __launch_bounds__(256) void out_proj(
    const _Float16* __restrict__ A, const _Float16* __restrict__ W,
    const float* __restrict__ bo, const float* __restrict__ residual,
    float* __restrict__ out) {
  __shared__ __align__(32) _Float16 Al[2][TM][TK];
  __shared__ __align__(32) _Float16 Wt[2][TK][TN];

  const int tid  = threadIdx.x;
  const int wave = tid >> 5;
  const int lane = tid & 31;
  const int m0 = blockIdx.x * TM;
  const int n0 = blockIdx.y * TN;
  const int K = CCH, Nout = CCH;

  const int arow = lane & 15;
  const int kb   = (lane >> 4) * 8;
  const int wn = tid >> 2;
  const int wkc = (tid & 3) * 8;

  v8f acc[2][4] = {};
  const int NIT = K / TK;

  {
    v8h wr = *(const v8h*)&W[(size_t)(n0 + wn) * K + wkc];
    if (wave == 0)
      tdm_load_2d_f16(lds_off(&Al[0][0][0]), &A[(size_t)m0 * K], TK, TM, K);
#pragma unroll
    for (int q = 0; q < 8; ++q) Wt[0][wkc + q][wn] = wr[q];
    if (wave == 0) __builtin_amdgcn_s_wait_tensorcnt(0);
  }
  __syncthreads();

  for (int it = 0; it < NIT; ++it) {
    const int cur = it & 1, nxt = cur ^ 1;
    const bool more = (it + 1) < NIT;
    const int kn = (it + 1) * TK;

    v8h wrn;
    if (more) {
      wrn = *(const v8h*)&W[(size_t)(n0 + wn) * K + kn + wkc];
      if (wave == 0)
        tdm_load_2d_f16(lds_off(&Al[nxt][0][0]), &A[(size_t)m0 * K + kn],
                        TK, TM, K);
    }

    union { v16h v; v8h h[2]; } af[2];
#pragma unroll
    for (int f = 0; f < 2; ++f) {
      af[f].h[0] = *(const v8h*)&Al[cur][wave * 32 + f * 16 + arow][kb];
      af[f].h[1] = *(const v8h*)&Al[cur][wave * 32 + f * 16 + arow][kb + 16];
    }
    v16h bf[4];
#pragma unroll
    for (int c = 0; c < 4; ++c)
#pragma unroll
      for (int s = 0; s < 16; ++s)
        bf[c][s] = Wt[cur][kb + (s & 7) + ((s >> 3) << 4)][c * 16 + arow];
#pragma unroll
    for (int f = 0; f < 2; ++f)
#pragma unroll
      for (int c = 0; c < 4; ++c)
        acc[f][c] = wmma_f16(af[f].v, bf[c], acc[f][c]);

    if (more) {
#pragma unroll
      for (int q = 0; q < 8; ++q) Wt[nxt][wkc + q][wn] = wrn[q];
      if (wave == 0) __builtin_amdgcn_s_wait_tensorcnt(0);
    }
    __syncthreads();
  }

  const int rb = m0 + wave * 32 + ((lane >> 4) << 3);
  const int cb = lane & 15;
#pragma unroll
  for (int f = 0; f < 2; ++f)
#pragma unroll
    for (int c = 0; c < 4; ++c) {
      int col = n0 + c * 16 + cb;
      float b = bo[col];
#pragma unroll
      for (int r = 0; r < 8; ++r) {
        int row = rb + f * 16 + r;
        out[(size_t)col * NTOK + row] =
            acc[f][c][r] + b + residual[(size_t)col * NTOK + row];
      }
    }
}

// ---------------------------------------------------------------------------
// launch
// ---------------------------------------------------------------------------
extern "C" void kernel_launch(void* const* d_in, const int* in_sizes, int n_in,
                              void* d_out, int out_size, void* d_ws, size_t ws_size,
                              hipStream_t stream) {
  (void)in_sizes; (void)n_in; (void)out_size; (void)ws_size;
  const float* x     = (const float*)d_in[0];
  const float* gamma = (const float*)d_in[1];
  const float* beta  = (const float*)d_in[2];
  const float* wq    = (const float*)d_in[3];
  const float* wk    = (const float*)d_in[4];
  const float* wv    = (const float*)d_in[5];
  const float* wo    = (const float*)d_in[6];
  const float* bo    = (const float*)d_in[7];
  float* out = (float*)d_out;

  const size_t WSZ = (size_t)CCH * CCH;
  const size_t XSZ = (size_t)NTOK * CCH;
  _Float16* ws   = (_Float16*)d_ws;
  _Float16* wq16 = ws;
  _Float16* wk16 = wq16 + WSZ;
  _Float16* wv16 = wk16 + WSZ;
  _Float16* wo16 = wv16 + WSZ;
  _Float16* Xn   = wo16 + WSZ;
  _Float16* Qb   = Xn + XSZ;
  _Float16* Kb   = Qb + XSZ;
  _Float16* Vb   = Kb + XSZ;
  _Float16* Ob   = Vb + XSZ;

  cvt_f32_to_f16<<<(int)(WSZ / 256), 256, 0, stream>>>(wq, wq16, (int)WSZ);
  cvt_f32_to_f16<<<(int)(WSZ / 256), 256, 0, stream>>>(wk, wk16, (int)WSZ);
  cvt_f32_to_f16<<<(int)(WSZ / 256), 256, 0, stream>>>(wv, wv16, (int)WSZ);
  cvt_f32_to_f16<<<(int)(WSZ / 256), 256, 0, stream>>>(wo, wo16, (int)WSZ);

  groupnorm_t16<<<NGRP, 256, 0, stream>>>(x, gamma, beta, Xn);

  dim3 gg(NTOK / TM, CCH / TN);   // (16, 8)
  gemm_xwT<<<gg, 256, 0, stream>>>(Xn, wq16, Qb, NTOK, CCH, CCH);
  gemm_xwT<<<gg, 256, 0, stream>>>(Xn, wk16, Kb, NTOK, CCH, CCH);
  gemm_xwT<<<gg, 256, 0, stream>>>(Xn, wv16, Vb, NTOK, CCH, CCH);

  dim3 ga(NTOK / 64, NHEAD);      // (64, 8)
  flash_attn<<<ga, 128, 0, stream>>>(Qb, Kb, Vb, Ob);

  out_proj<<<gg, 256, 0, stream>>>(Ob, wo16, bo, x, out);
}
